// CL_spatial_global_77077483094318
// MI455X (gfx1250) — matrix operand order repaired and checked by reading, not compiled, not run on hardware
//
#include <hip/hip_runtime.h>
#include <hip/hip_bf16.h>

typedef __attribute__((ext_vector_type(16))) _Float16 v16h;
typedef __attribute__((ext_vector_type(8)))  float    v8f;
typedef __attribute__((ext_vector_type(4)))  float    v4f;

static constexpr int B_ = 16;
static constexpr int C_ = 256;
static constexpr int N_ = 1024;   // H*W
static constexpr int P_ = 64;

// ---------------------------------------------------------------------------
// K1: projection  h[b,n,c] = elu(z_row @ W1^T + b1) @ W2^T + b2 ; rn[b,n]=||h row||
// ---------------------------------------------------------------------------
__global__ __launch_bounds__(256) void proj_kernel(
    const float* __restrict__ z, const float* __restrict__ w1, const float* __restrict__ b1,
    const float* __restrict__ w2, const float* __restrict__ b2,
    float* __restrict__ h, float* __restrict__ rn)
{
    const int b = blockIdx.y;
    const int n = blockIdx.x * 256 + threadIdx.x;
    const float* zb = z + (size_t)b * C_ * N_;

    float acc[P_];
#pragma unroll
    for (int p = 0; p < P_; ++p) acc[p] = b1[p];

#pragma unroll 1
    for (int c = 0; c < C_; ++c) {
        float zv = zb[(size_t)c * N_ + n];           // coalesced across threads
#pragma unroll
        for (int p = 0; p < P_; ++p)
            acc[p] = fmaf(zv, w1[p * C_ + c], acc[p]);   // uniform (scalar) weight loads
    }
#pragma unroll
    for (int p = 0; p < P_; ++p) {
        float x = acc[p];
        acc[p] = x > 0.0f ? x : expm1f(x);           // elu, alpha=1
    }

    float* hrow = h + ((size_t)b * N_ + n) * C_;
    float sq = 0.0f;
#pragma unroll 1
    for (int c = 0; c < C_; ++c) {
        float s = b2[c];
#pragma unroll
        for (int p = 0; p < P_; ++p)
            s = fmaf(acc[p], w2[c * P_ + p], s);
        hrow[c] = s;
        sq = fmaf(s, s, sq);
    }
    rn[b * N_ + n] = sqrtf(sq);
}

// ---------------------------------------------------------------------------
// K2a: cn[n] = max(sqrt(sum_b rn[b,n]^2), 1e-12)
// ---------------------------------------------------------------------------
__global__ __launch_bounds__(256) void colnorm_kernel(const float* __restrict__ rn,
                                                      float* __restrict__ cn)
{
    int n = blockIdx.x * 256 + threadIdx.x;
    float s = 0.0f;
#pragma unroll
    for (int b = 0; b < B_; ++b) { float r = rn[b * N_ + n]; s = fmaf(r, r, s); }
    cn[n] = fmaxf(sqrtf(s), 1e-12f);
}

// ---------------------------------------------------------------------------
// K2: Gt[n][c] = (sum_b h[b,n,c]) / cn[n]  stored as f16 (B-operand feed)
// ---------------------------------------------------------------------------
__global__ __launch_bounds__(256) void gmat_kernel(const float* __restrict__ h,
                                                   const float* __restrict__ cn,
                                                   _Float16* __restrict__ Gt)
{
    int gid = blockIdx.x * 256 + threadIdx.x;   // over N*C, c fastest -> coalesced
    int n = gid / C_, c = gid % C_;
    float s = 0.0f;
#pragma unroll
    for (int b = 0; b < B_; ++b) s += h[((size_t)b * N_ + n) * C_ + c];
    Gt[(size_t)n * C_ + c] = (_Float16)(s / cn[n]);
}

// ---------------------------------------------------------------------------
// K3: per-row dots of normalized rows (exact f32 diagonals)
// ---------------------------------------------------------------------------
__global__ __launch_bounds__(256) void dots_kernel(
    const float* __restrict__ h1, const float* __restrict__ h2,
    const float* __restrict__ rn1, const float* __restrict__ rn2,
    float* __restrict__ idot, float* __restrict__ sd1, float* __restrict__ sd2)
{
    int row = blockIdx.x * 256 + threadIdx.x;   // over B*N
    const float* r1 = h1 + (size_t)row * C_;
    const float* r2 = h2 + (size_t)row * C_;
    float i1 = 1.0f / fmaxf(rn1[row], 1e-12f);
    float i2 = 1.0f / fmaxf(rn2[row], 1e-12f);
    float dd = 0.0f, s1 = 0.0f, s2 = 0.0f;
#pragma unroll 4
    for (int c = 0; c < C_; ++c) {
        float v1 = r1[c] * i1, v2 = r2[c] * i2;
        dd = fmaf(v1, v2, dd); s1 = fmaf(v1, v1, s1); s2 = fmaf(v2, v2, s2);
    }
    idot[row] = dd; sd1[row] = s1; sd2[row] = s2;
}

// ---------------------------------------------------------------------------
// K4: WMMA sim-matrix + exp-rowsum + per-row loss.
// One wave per (32-row i-block, b, direction). Two A tiles (32 x 256 f16) in
// regs. B (g^T) streams as K=128 half-tile jobs through two 4-fragment
// register buffers (load half-job t+1 while the 8 WMMAs of half-job t run).
// The two exp-sum streams share accumulators (loss needs only S1+S2), which
// keeps total VGPR usage under 256 (no s_set_vgpr_msb paging).
// ---------------------------------------------------------------------------
__global__ __launch_bounds__(32) void simexp_kernel(
    const float* __restrict__ h1, const float* __restrict__ h2,
    const float* __restrict__ rn1, const float* __restrict__ rn2,
    const _Float16* __restrict__ G1, const _Float16* __restrict__ G2,
    const float* __restrict__ idot, const float* __restrict__ sd1, const float* __restrict__ sd2,
    float* __restrict__ L)
{
    const int lane = threadIdx.x;
    const int it = blockIdx.x;      // i-block: rows it*32 .. it*32+31
    const int b  = blockIdx.y;
    const int d  = blockIdx.z;      // direction

    const float*    hA     = d == 0 ? h1  : h2;
    const float*    rnA    = d == 0 ? rn1 : rn2;
    const _Float16* Gintra = d == 0 ? G1  : G2;
    const _Float16* Ginter = d == 0 ? G2  : G1;
    const float*    sd     = d == 0 ? sd1 : sd2;

    const int m  = lane & 15;
    const int hi = lane >> 4;
    const int rowL = it * 32 + m;          // low A tile rows
    const float* arowL = hA + ((size_t)b * N_ + rowL) * C_;
    const float* arowH = arowL + (size_t)16 * C_;
    const float invL = 1.0f / fmaxf(rnA[b * N_ + rowL], 1e-12f);
    const float invH = 1.0f / fmaxf(rnA[b * N_ + rowL + 16], 1e-12f);

    // A fragments: contiguous K order (matches B). lanes 0-15: K{0..15};
    // lanes 16-31: K{16..31} -> per-lane contiguous chunk offset hi*16.
    v16h aL[8], aH[8];
#pragma unroll
    for (int kk = 0; kk < 8; ++kk) {
        const int c0 = kk * 32 + hi * 16;
        {
            v4f x0 = *(const v4f*)(arowL + c0);
            v4f x1 = *(const v4f*)(arowL + c0 + 4);
            v4f y0 = *(const v4f*)(arowL + c0 + 8);
            v4f y1 = *(const v4f*)(arowL + c0 + 12);
            v16h f;
#pragma unroll
            for (int j = 0; j < 4; ++j) {
                f[j]      = (_Float16)(x0[j] * invL);
                f[4 + j]  = (_Float16)(x1[j] * invL);
                f[8 + j]  = (_Float16)(y0[j] * invL);
                f[12 + j] = (_Float16)(y1[j] * invL);
            }
            aL[kk] = f;
        }
        {
            v4f x0 = *(const v4f*)(arowH + c0);
            v4f x1 = *(const v4f*)(arowH + c0 + 4);
            v4f y0 = *(const v4f*)(arowH + c0 + 8);
            v4f y1 = *(const v4f*)(arowH + c0 + 12);
            v16h f;
#pragma unroll
            for (int j = 0; j < 4; ++j) {
                f[j]      = (_Float16)(x0[j] * invH);
                f[4 + j]  = (_Float16)(x1[j] * invH);
                f[8 + j]  = (_Float16)(y0[j] * invH);
                f[12 + j] = (_Float16)(y1[j] * invH);
            }
            aH[kk] = f;
        }
    }

    // combined exp-sums over BOTH g-streams (denominator only needs S1+S2)
    float psL[8], psH[8];
#pragma unroll
    for (int r = 0; r < 8; ++r) { psL[r] = 0.0f; psH[r] = 0.0f; }

    const float scale = 0.625f;          // 1/(tau*sqrt(B)) = 1/(0.4*4)
    const int nB = lane & 15;            // B-fragment column within tile
    const int kh = hi * 16;              // B-fragment K half
    const _Float16* gbase1 = Gintra + (size_t)nB * C_ + kh;
    const _Float16* gbase2 = Ginter + (size_t)nB * C_ + kh;

    // half-tile = K range of 128 (fragments kk..kk+3); 4 x v16h per buffer
    auto loadHalf = [](const _Float16* p, v16h (&buf)[4]) {
#pragma unroll
        for (int i = 0; i < 4; ++i) buf[i] = *(const v16h*)(p + i * 32);
    };

    v16h buf0[4], buf1[4];
    loadHalf(gbase1, buf0);              // (G_intra, jt=0, k-half 0)

#pragma unroll 1
    for (int jt = 0; jt < 64; ++jt) {
        const _Float16* p1 = gbase1 + (size_t)jt * 16 * C_;
        const _Float16* p2 = gbase2 + (size_t)jt * 16 * C_;

        v8f accL, accH;
        // ---- G_intra tile: halves in buf0 (ready) / buf1 (loading) ----
        loadHalf(p1 + 128, buf1);
        accL = {}; accH = {};
#pragma unroll
        for (int i = 0; i < 4; ++i) {
            accL = __builtin_amdgcn_wmma_f32_16x16x32_f16(
                       false, aL[i], false, buf0[i], (short)0, accL, false, false);
            accH = __builtin_amdgcn_wmma_f32_16x16x32_f16(
                       false, aH[i], false, buf0[i], (short)0, accH, false, false);
        }
        loadHalf(p2, buf0);              // prefetch (G_inter, k-half 0)
#pragma unroll
        for (int i = 0; i < 4; ++i) {
            accL = __builtin_amdgcn_wmma_f32_16x16x32_f16(
                       false, aL[4 + i], false, buf1[i], (short)0, accL, false, false);
            accH = __builtin_amdgcn_wmma_f32_16x16x32_f16(
                       false, aH[4 + i], false, buf1[i], (short)0, accH, false, false);
        }
#pragma unroll
        for (int r = 0; r < 8; ++r) {
            psL[r] += __expf(accL[r] * scale);
            psH[r] += __expf(accH[r] * scale);
        }

        // ---- G_inter tile: halves in buf0 (ready) / buf1 (loading) ----
        loadHalf(p2 + 128, buf1);
        accL = {}; accH = {};
#pragma unroll
        for (int i = 0; i < 4; ++i) {
            accL = __builtin_amdgcn_wmma_f32_16x16x32_f16(
                       false, aL[i], false, buf0[i], (short)0, accL, false, false);
            accH = __builtin_amdgcn_wmma_f32_16x16x32_f16(
                       false, aH[i], false, buf0[i], (short)0, accH, false, false);
        }
        if (jt < 63) loadHalf(p1 + 16 * C_, buf0);   // prefetch next (G_intra, k0)
#pragma unroll
        for (int i = 0; i < 4; ++i) {
            accL = __builtin_amdgcn_wmma_f32_16x16x32_f16(
                       false, aL[4 + i], false, buf1[i], (short)0, accL, false, false);
            accH = __builtin_amdgcn_wmma_f32_16x16x32_f16(
                       false, aH[4 + i], false, buf1[i], (short)0, accH, false, false);
        }
#pragma unroll
        for (int r = 0; r < 8; ++r) {
            psL[r] += __expf(accL[r] * scale);
            psH[r] += __expf(accH[r] * scale);
        }
    }

    // reduce over the 16 lanes of each half (N dimension of the D tiles)
#pragma unroll
    for (int r = 0; r < 8; ++r) {
        float a0 = psL[r], a1 = psH[r];
#pragma unroll
        for (int mk = 1; mk < 16; mk <<= 1) {
            a0 += __shfl_xor(a0, mk, 32);
            a1 += __shfl_xor(a1, mk, 32);
        }
        psL[r] = a0; psH[r] = a1;
    }

    if ((lane & 15) == 0) {
        const float itau = 2.5f;         // 1/tau
#pragma unroll
        for (int r = 0; r < 8; ++r) {
            {   // low tile: D layout VGPR r -> M = hi*8 + r
                int row = it * 32 + hi * 8 + r;
                int idx = b * N_ + row;
                float denom = psL[r] - __expf(sd[idx] * itau);
                L[(size_t)d * B_ * N_ + idx] = __logf(denom) - idot[idx] * itau;
            }
            {   // high tile
                int row = it * 32 + 16 + hi * 8 + r;
                int idx = b * N_ + row;
                float denom = psH[r] - __expf(sd[idx] * itau);
                L[(size_t)d * B_ * N_ + idx] = __logf(denom) - idot[idx] * itau;
            }
        }
    }
}

// ---------------------------------------------------------------------------
// K5: deterministic reduction to the scalar mean: ((l1+l2)*0.5).mean()
// ---------------------------------------------------------------------------
__global__ __launch_bounds__(256) void reduce_kernel(const float* __restrict__ L,
                                                     float* __restrict__ out)
{
    __shared__ float sm[256];
    float s = 0.0f;
    for (int i = threadIdx.x; i < 2 * B_ * N_; i += 256) s += L[i];
    sm[threadIdx.x] = s;
    __syncthreads();
    for (int k = 128; k > 0; k >>= 1) {
        if (threadIdx.x < k) sm[threadIdx.x] += sm[threadIdx.x + k];
        __syncthreads();
    }
    if (threadIdx.x == 0) out[0] = sm[0] * (0.5f / (float)(B_ * N_));
}

// ---------------------------------------------------------------------------
extern "C" void kernel_launch(void* const* d_in, const int* in_sizes, int n_in,
                              void* d_out, int out_size, void* d_ws, size_t ws_size,
                              hipStream_t stream)
{
    const float* z1 = (const float*)d_in[0];
    const float* z2 = (const float*)d_in[1];
    const float* w1 = (const float*)d_in[2];
    const float* b1 = (const float*)d_in[3];
    const float* w2 = (const float*)d_in[4];
    const float* b2 = (const float*)d_in[5];
    float* out = (float*)d_out;

    char* p = (char*)d_ws;
    float*    h1  = (float*)p;    p += (size_t)B_ * N_ * C_ * 4;
    float*    h2  = (float*)p;    p += (size_t)B_ * N_ * C_ * 4;
    _Float16* G1  = (_Float16*)p; p += (size_t)N_ * C_ * 2;
    _Float16* G2  = (_Float16*)p; p += (size_t)N_ * C_ * 2;
    float*    rn1 = (float*)p;    p += (size_t)B_ * N_ * 4;
    float*    rn2 = (float*)p;    p += (size_t)B_ * N_ * 4;
    float*    cn1 = (float*)p;    p += (size_t)N_ * 4;
    float*    cn2 = (float*)p;    p += (size_t)N_ * 4;
    float*    idt = (float*)p;    p += (size_t)B_ * N_ * 4;
    float*    sd1 = (float*)p;    p += (size_t)B_ * N_ * 4;
    float*    sd2 = (float*)p;    p += (size_t)B_ * N_ * 4;
    float*    L   = (float*)p;    p += (size_t)2 * B_ * N_ * 4;

    proj_kernel<<<dim3(N_ / 256, B_), 256, 0, stream>>>(z1, w1, b1, w2, b2, h1, rn1);
    proj_kernel<<<dim3(N_ / 256, B_), 256, 0, stream>>>(z2, w1, b1, w2, b2, h2, rn2);
    colnorm_kernel<<<dim3(N_ / 256), 256, 0, stream>>>(rn1, cn1);
    colnorm_kernel<<<dim3(N_ / 256), 256, 0, stream>>>(rn2, cn2);
    gmat_kernel<<<dim3(N_ * C_ / 256), 256, 0, stream>>>(h1, cn1, G1);
    gmat_kernel<<<dim3(N_ * C_ / 256), 256, 0, stream>>>(h2, cn2, G2);
    dots_kernel<<<dim3(B_ * N_ / 256), 256, 0, stream>>>(h1, h2, rn1, rn2, idt, sd1, sd2);
    simexp_kernel<<<dim3(N_ / 32, B_, 2), 32, 0, stream>>>(h1, h2, rn1, rn2, G1, G2,
                                                           idt, sd1, sd2, L);
    reduce_kernel<<<1, 256, 0, stream>>>(L, out);
}
